// SequentialLayer_14139032338967
// MI455X (gfx1250) — matrix operand order repaired and checked by reading, not compiled, not run on hardware
//
#include <hip/hip_runtime.h>
#include <hip/hip_bf16.h>

typedef __attribute__((ext_vector_type(16))) _Float16 v16h;
typedef __attribute__((ext_vector_type(8)))  float    v8f;
typedef __attribute__((ext_vector_type(4)))  float    v4f;

#define S_LEN 1024
#define B_LEN 512
#define F_LEN 128
#define H_LEN 3
#define SB    (S_LEN * B_LEN)

// ---------------------------------------------------------------------------
// Phase 1: xp[m][n] = sum_k x[m][k] * W_ih[n][k] + b_ih[n]   (m = s*B+b)
// 16-row tiles, K=128 as 4x v_wmma_f32_16x16x32_f16, f32 accumulate.
// Memory bound: reads 256 MB of x -> ~11 us at 23.3 TB/s.
// ---------------------------------------------------------------------------
__global__ __launch_bounds__(256) void xproj_wmma_kernel(
    const float* __restrict__ x,      // (SB, 128)
    const float* __restrict__ W_ih,   // (3, 128)
    const float* __restrict__ b_ih,   // (3)
    float* __restrict__ xp,           // (SB, 3)
    int M)
{
  const int lane = threadIdx.x & 31;
  const int wave = threadIdx.x >> 5;
  const int half = lane >> 4;          // 0: lanes 0-15, 1: lanes 16-31
  const int l    = lane & 15;
  const int n    = l;                  // output column owned by this lane

  // B fragments: W_ih^T zero-padded to 16 columns. Tile-invariant -> build once.
  // B 32x16 f16 layout: N=lane&15, elem e<8 <-> K=8*half+e; e>=8 <-> K=16+8*half+(e-8).
  v16h bfrag[4];
  #pragma unroll
  for (int c = 0; c < 4; ++c) {
    v16h bf;
    if (n < H_LEN) {
      const float* wrow = W_ih + n * F_LEN + c * 32 + half * 8;
      #pragma unroll
      for (int e = 0; e < 16; ++e) bf[e] = (_Float16)wrow[e < 8 ? e : e + 8];
    } else {
      #pragma unroll
      for (int e = 0; e < 16; ++e) bf[e] = (_Float16)0.0f;
    }
    bfrag[c] = bf;
  }
  const float bias = (n < H_LEN) ? b_ih[n] : 0.0f;

  const int nTiles = M >> 4;
  const int aoff   = half * 8;         // A lanes 16-31 hold K offset +8
  for (int tile = blockIdx.x * 8 + wave; tile < nTiles; tile += gridDim.x * 8) {
    const int m0 = tile << 4;
    const float* xrow = x + (size_t)(m0 + l) * F_LEN;

    v8f acc = {};
    #pragma unroll
    for (int c = 0; c < 4; ++c) {
      const float* p = xrow + c * 32 + aoff;
      v4f f0 = *(const v4f*)(p + 0);    // K = kb+aoff+0..3
      v4f f1 = *(const v4f*)(p + 4);    // K = kb+aoff+4..7
      v4f f2 = *(const v4f*)(p + 16);   // K = kb+aoff+16..19
      v4f f3 = *(const v4f*)(p + 20);   // K = kb+aoff+20..23
      v16h af;
      #pragma unroll
      for (int e = 0; e < 4; ++e) {
        af[e]      = (_Float16)f0[e];
        af[e + 4]  = (_Float16)f1[e];
        af[e + 8]  = (_Float16)f2[e];
        af[e + 12] = (_Float16)f3[e];
      }
      acc = __builtin_amdgcn_wmma_f32_16x16x32_f16(
          /*neg_a=*/false, af, /*neg_b=*/false, bfrag[c],
          /*c_mod=*/(short)0, acc, /*reuse_a=*/false, /*reuse_b=*/false);
    }

    // D layout: VGPR r -> row r (lanes 0-15) / row 8+r (lanes 16-31), col = lane&15.
    if (n < H_LEN) {
      #pragma unroll
      for (int r = 0; r < 8; ++r) {
        int row = m0 + half * 8 + r;
        xp[(size_t)row * H_LEN + n] = acc[r] + bias;
      }
    }
  }
}

// ---------------------------------------------------------------------------
// Fast tanh: hardware v_tanh_f32 on gfx1250 if available, else exp2-based.
// ---------------------------------------------------------------------------
__device__ __forceinline__ float fast_tanh(float xx) {
#if __has_builtin(__builtin_amdgcn_tanhf)
  return __builtin_amdgcn_tanhf(xx);
#else
  float xc = fminf(fmaxf(xx, -10.0f), 10.0f);
  float e  = __builtin_amdgcn_exp2f(-2.8853900817779268f * xc); // exp(-2x)
  return (1.0f - e) * __builtin_amdgcn_rcpf(1.0f + e);
#endif
}

// ---------------------------------------------------------------------------
// Phase 2a: serial scan. h = tanh(xp[s] + h @ W_hh^T + b_hh). One lane per b.
// Depth-8 register prefetch pipeline hides L2 latency of xp (L2-hot, 6 MB).
// Writes all step outputs to out_h and final h to hT.
// ---------------------------------------------------------------------------
__global__ __launch_bounds__(32) void rnn_scan_kernel(
    const float* __restrict__ xp,     // (S, B, 3)
    const float* __restrict__ W_hh,   // (3, 3)
    const float* __restrict__ b_hh,   // (3)
    float* __restrict__ out_h,        // (S, B, 3)
    float* __restrict__ hT)           // (B, 3)
{
  const int b = blockIdx.x * 32 + threadIdx.x;
  if (b >= B_LEN) return;

  const float w00 = W_hh[0], w01 = W_hh[1], w02 = W_hh[2];
  const float w10 = W_hh[3], w11 = W_hh[4], w12 = W_hh[5];
  const float w20 = W_hh[6], w21 = W_hh[7], w22 = W_hh[8];
  const float bb0 = b_hh[0], bb1 = b_hh[1], bb2 = b_hh[2];

  float h0 = 0.0f, h1 = 0.0f, h2 = 0.0f;

  constexpr int PF = 8;
  float px[PF], py[PF], pz[PF];
  #pragma unroll
  for (int i = 0; i < PF; ++i) {
    size_t base = ((size_t)i * B_LEN + b) * H_LEN;
    px[i] = xp[base + 0]; py[i] = xp[base + 1]; pz[i] = xp[base + 2];
  }

  for (int s0 = 0; s0 < S_LEN; s0 += PF) {
    #pragma unroll
    for (int i = 0; i < PF; ++i) {
      const int s = s0 + i;
      const float c0 = px[i] + bb0;
      const float c1 = py[i] + bb1;
      const float c2 = pz[i] + bb2;

      // prefetch step s+PF before entering the dependent chain
      const int sp = s + PF;
      if (sp < S_LEN) {
        size_t pb = ((size_t)sp * B_LEN + b) * H_LEN;
        px[i] = xp[pb + 0]; py[i] = xp[pb + 1]; pz[i] = xp[pb + 2];
      }

      // serial chain: 3 FMAs (ILP-3 across j) + tanh
      const float a0 = fmaf(h0, w00, fmaf(h1, w01, fmaf(h2, w02, c0)));
      const float a1 = fmaf(h0, w10, fmaf(h1, w11, fmaf(h2, w12, c1)));
      const float a2 = fmaf(h0, w20, fmaf(h1, w21, fmaf(h2, w22, c2)));
      h0 = fast_tanh(a0);
      h1 = fast_tanh(a1);
      h2 = fast_tanh(a2);

      size_t ob = ((size_t)s * B_LEN + b) * H_LEN;
      out_h[ob + 0] = h0; out_h[ob + 1] = h1; out_h[ob + 2] = h2;
    }
  }

  hT[(size_t)b * H_LEN + 0] = h0;
  hT[(size_t)b * H_LEN + 1] = h1;
  hT[(size_t)b * H_LEN + 2] = h2;
}

// ---------------------------------------------------------------------------
// Phase 2b: gumbel hard-argmax. Numerically y = one_hot(argmax(h+g)), so
// delta = signal[argmax]. Fully parallel, ~14 MB of traffic -> <1 us.
// ---------------------------------------------------------------------------
__global__ __launch_bounds__(256) void gumbel_delta_kernel(
    const float* __restrict__ out_h,  // (SB, 3)
    const float* __restrict__ gumbel, // (SB, 3)
    const float* __restrict__ signal, // (3)
    float* __restrict__ delta,        // (SB)
    int M)
{
  const int m = blockIdx.x * 256 + threadIdx.x;
  if (m >= M) return;
  const size_t t = (size_t)m * H_LEN;
  const float l0 = out_h[t + 0] + gumbel[t + 0];
  const float l1 = out_h[t + 1] + gumbel[t + 1];
  const float l2 = out_h[t + 2] + gumbel[t + 2];
  // first-occurrence argmax, matching jnp.argmax
  int idx = 0; float best = l0;
  if (l1 > best) { best = l1; idx = 1; }
  if (l2 > best) { idx = 2; }
  const float s0 = signal[0], s1 = signal[1], s2 = signal[2];
  delta[m] = (idx == 0) ? s0 : ((idx == 1) ? s1 : s2);
}

// ---------------------------------------------------------------------------
extern "C" void kernel_launch(void* const* d_in, const int* in_sizes, int n_in,
                              void* d_out, int out_size, void* d_ws, size_t ws_size,
                              hipStream_t stream) {
  const float* x      = (const float*)d_in[0];   // (S,B,F)
  const float* gumbel = (const float*)d_in[1];   // (S,B,H)
  const float* W_ih   = (const float*)d_in[2];   // (H,F)
  const float* W_hh   = (const float*)d_in[3];   // (H,H)
  const float* b_ih   = (const float*)d_in[4];   // (H)
  const float* b_hh   = (const float*)d_in[5];   // (H)
  const float* signal = (const float*)d_in[6];   // (H,1)

  float* delta = (float*)d_out;                  // first S*B floats
  float* hT    = delta + SB;                     // then B*H floats

  // workspace: xp (S*B*H f32) + out_h (S*B*H f32) = 12 MB
  float* xp    = (float*)d_ws;
  float* out_h = xp + (size_t)SB * H_LEN;

  // Phase 1: WMMA input projection (memory-bound on 256 MB of x)
  xproj_wmma_kernel<<<1024, 256, 0, stream>>>(x, W_ih, b_ih, xp, SB);

  // Phase 2a: serial RNN scan, 512 lanes across 16 single-wave workgroups
  rnn_scan_kernel<<<16, 32, 0, stream>>>(xp, W_hh, b_hh, out_h, hT);

  // Phase 2b: parallel gumbel argmax -> delta
  gumbel_delta_kernel<<<(SB + 255) / 256, 256, 0, stream>>>(out_h, gumbel, signal, delta, SB);

  (void)in_sizes; (void)n_in; (void)out_size; (void)ws_size;
}
